// MultiheadAttention_65755949302049
// MI455X (gfx1250) — matrix operand order repaired and checked by reading, not compiled, not run on hardware
//
#include <hip/hip_runtime.h>
#include <stdint.h>

// ---------------------------------------------------------------------------
// MultiheadAttention (modality-masked) forward for gfx1250 (MI455X, wave32).
// bf16 WMMA (v_wmma_f32_16x16x32_bf16) for all three GEMM stages, fp32 softmax.
// Dense GEMMs register-blocked 32x32 per wave (4 accumulators) for 2x L2
// arithmetic intensity.
// ---------------------------------------------------------------------------

typedef unsigned short bf16_t;   // raw bf16 bits for storage
typedef __attribute__((ext_vector_type(16))) __bf16 v16bf;
typedef __attribute__((ext_vector_type(8)))  float  v8f;

#define T_LEN 1536
#define EMB   1024
#define NB    4
#define NH    16
#define HD    64
#define MROWS 6144          // T_LEN * NB
#define ATTN_SCALE 0.125f   // 64^-0.5

union FragBF { v16bf v; uint32_t u[8]; };
union FragF  { v8f   v; float    f[8]; };

__device__ __forceinline__ bf16_t f2bf(float x) {
  uint32_t u = __builtin_bit_cast(uint32_t, x);
  u += 0x7FFFu + ((u >> 16) & 1u);          // round-to-nearest-even
  return (bf16_t)(u >> 16);
}

// A-matrix fragment, 16x32 bf16 (MxK). base -> element (row0, k0), lda in elems.
// ISA layout: lanes 0-15 row=lane, kb=0; lanes 16-31 row=lane-16, kb=8.
// VGPR v<4 -> K = kb + 2v,2v+1 ; VGPR v>=4 -> K = kb+16 + 2(v-4), +1.
// (VGPR0-3 and VGPR4-7 are byte-contiguous per lane -> compiler emits 2x b128.)
__device__ __forceinline__ v16bf load_a(const bf16_t* base, int lda, int lane) {
  FragBF f;
  int r  = lane & 15;
  int kb = (lane >> 4) << 3;
  const bf16_t* p = base + r * lda + kb;
  #pragma unroll
  for (int v = 0; v < 8; ++v) {
    int k0 = (v < 4) ? (2 * v) : (16 + 2 * (v - 4));
    f.u[v] = *(const uint32_t*)(p + k0);
  }
  return f.v;
}

// B-matrix fragment, 32x16 bf16 (KxN), stored N-major / K-contiguous:
// element (k,n) at base[n*ldb + k]. Lanes 0-15: n=lane, kb=0; lanes 16-31: kb=8 (same K order as load_a).
__device__ __forceinline__ v16bf load_b(const bf16_t* base, int ldb, int lane) {
  FragBF f;
  int n  = lane & 15;
  int kb = (lane >> 4) << 3;
  const bf16_t* p = base + n * ldb + kb;
  #pragma unroll
  for (int v = 0; v < 8; ++v) f.u[v] = *(const uint32_t*)(p + ((v < 4) ? (2 * v) : (16 + 2 * (v - 4))));
  return f.v;
}

__device__ __forceinline__ v8f wmma_bf16(v16bf a, v16bf b, v8f c) {
  return __builtin_amdgcn_wmma_f32_16x16x32_bf16(
      /*neg_a=*/false, a, /*neg_b=*/false, b,
      /*c_mod=*/(short)0, c, /*reuse_a=*/false, /*reuse_b=*/false);
}

__device__ __forceinline__ bool is_masked(int i, int j) {
  const int l = 512, la = 1024;
  return ((i < l)  && (j >= l) && (j < la))
      || ((i >= l) && (i < la) && (j < l))
      || ((i < la) && (j >= la + 1))
      || ((i >= la + 1) && (j < la))
      || ((i >= la) && (j >= la) && (j > i));
}

// ---------------------------------------------------------------------------
// Kernel 0: fp32 -> bf16 staging conversions
// ---------------------------------------------------------------------------
__global__ __launch_bounds__(256) void k_convert(
    const float* __restrict__ q, const float* __restrict__ w,
    const float* __restrict__ ow,
    bf16_t* __restrict__ X, bf16_t* __restrict__ W, bf16_t* __restrict__ OW) {
  const int NX = MROWS * EMB;       // query flattened [T*B, E]
  const int NW = 3 * EMB * EMB;     // in_proj_weight [3E, E]
  const int NO = EMB * EMB;         // out_w [E, E]
  for (int i = blockIdx.x * blockDim.x + threadIdx.x; i < NX;
       i += gridDim.x * blockDim.x) {
    X[i] = f2bf(q[i]);
    if (i < NW) W[i]  = f2bf(w[i]);
    if (i < NO) OW[i] = f2bf(ow[i]);
  }
}

__global__ __launch_bounds__(256) void k_zero(float* __restrict__ p, int n) {
  for (int i = blockIdx.x * blockDim.x + threadIdx.x; i < n;
       i += gridDim.x * blockDim.x) p[i] = 0.0f;
}

// ---------------------------------------------------------------------------
// Kernel 1: fused QKV projection. qkv = X @ W^T + b. M=6144, N=3072, K=1024.
// 32x32 register block per wave (2x2 WMMA tiles). Epilogue scatters to
// Q/K [B,H,T,64] and V^T [B,H,64,T].
// ---------------------------------------------------------------------------
__global__ __launch_bounds__(256) void k_qkv(
    const bf16_t* __restrict__ X, const bf16_t* __restrict__ W,
    const float* __restrict__ bias,
    bf16_t* __restrict__ Q, bf16_t* __restrict__ K, bf16_t* __restrict__ V) {
  int lane = threadIdx.x & 31, wave = threadIdx.x >> 5;
  int tile = blockIdx.x * 8 + wave;           // 2304 blocks * 8 = 18432 tiles
  int mt = tile / 96, nt = tile % 96;         // 192 x 96 tiles of 32x32
  const bf16_t* A0 = X + (size_t)mt * 32 * EMB;
  const bf16_t* A1 = A0 + (size_t)16 * EMB;
  const bf16_t* B0 = W + (size_t)nt * 32 * EMB;  // row n of W is K-contiguous
  const bf16_t* B1 = B0 + (size_t)16 * EMB;
  v8f acc00 = {}, acc01 = {}, acc10 = {}, acc11 = {};
  for (int k = 0; k < EMB; k += 32) {
    v16bf a0 = load_a(A0 + k, EMB, lane);
    v16bf a1 = load_a(A1 + k, EMB, lane);
    v16bf b0 = load_b(B0 + k, EMB, lane);
    v16bf b1 = load_b(B1 + k, EMB, lane);
    acc00 = wmma_bf16(a0, b0, acc00);
    acc01 = wmma_bf16(a0, b1, acc01);
    acc10 = wmma_bf16(a1, b0, acc10);
    acc11 = wmma_bf16(a1, b1, acc11);
  }
  int n = lane & 15, hs = lane >> 4;
  #pragma unroll
  for (int si = 0; si < 2; ++si) {
    #pragma unroll
    for (int sj = 0; sj < 2; ++sj) {
      FragF f;
      f.v = si == 0 ? (sj == 0 ? acc00 : acc01) : (sj == 0 ? acc10 : acc11);
      int ng = nt * 32 + sj * 16 + n;
      int sec = ng >> 10, e = ng & 1023, h = e >> 6, d = e & 63;
      float bsv = bias[ng];
      size_t hstride = (size_t)NH * T_LEN * HD;   // per-batch stride
      #pragma unroll
      for (int v = 0; v < 8; ++v) {
        int m = mt * 32 + si * 16 + hs * 8 + v;   // row = t*B + b
        int t = m >> 2, b = m & 3;
        float val = f.f[v] + bsv;
        size_t base = (size_t)b * hstride + (size_t)h * T_LEN * HD;
        if (sec == 0)      Q[base + (size_t)t * HD + d] = f2bf(val);
        else if (sec == 1) K[base + (size_t)t * HD + d] = f2bf(val);
        else               V[base + (size_t)d * T_LEN + t] = f2bf(val);
      }
    }
  }
}

// ---------------------------------------------------------------------------
// Kernel 2: attention. One block per (b,h, 16-row tile): 6144 blocks, 8 waves.
// Scores in registers (wave owns 12 col-tiles; Q frags reused across all of
// them), fp32 softmax, P -> LDS bf16, P@V via WMMA (4 col tiles x 2 K-halves
// over the 8 waves), attn_avg accumulated across heads with f32 atomics.
// ---------------------------------------------------------------------------
__global__ __launch_bounds__(256) void k_attn(
    const bf16_t* __restrict__ Q, const bf16_t* __restrict__ K,
    const bf16_t* __restrict__ V, bf16_t* __restrict__ O,
    float* __restrict__ attn_avg) {
  __shared__ bf16_t Psh[16 * T_LEN];          // 48 KB probabilities (bf16)
  __shared__ float redmax[8][16];
  __shared__ float redsum[8][16];
  __shared__ float accred[4 * 8 * 32];        // PV partial tiles (4 KB)

  int lane = threadIdx.x & 31, wave = threadIdx.x >> 5;
  int n = lane & 15, hs = lane >> 4;
  int rt = blockIdx.x % 96;                   // 96 row tiles of 16
  int bh = blockIdx.x / 96;                   // b*16 + h
  int t0 = rt * 16;
  int bb = bh >> 4;

  const bf16_t* Qh = Q + (size_t)bh * T_LEN * HD;
  const bf16_t* Kh = K + (size_t)bh * T_LEN * HD;
  const bf16_t* Vh = V + (size_t)bh * HD * T_LEN;     // [64][T]

  // Q fragments for this row tile (K = 0..63), reused across all col tiles.
  v16bf qa0 = load_a(Qh + (size_t)t0 * HD, HD, lane);
  v16bf qa1 = load_a(Qh + (size_t)t0 * HD + 32, HD, lane);

  // ---- scores: S = scale * Q K^T + mask, kept in C-fragment registers ----
  float c[12][8];
  #pragma unroll
  for (int ct = 0; ct < 12; ++ct) {
    int s0 = (wave * 12 + ct) * 16;
    v8f acc = {};
    acc = wmma_bf16(qa0, load_b(Kh + (size_t)s0 * HD, HD, lane), acc);
    acc = wmma_bf16(qa1, load_b(Kh + (size_t)s0 * HD + 32, HD, lane), acc);
    FragF f; f.v = acc;
    #pragma unroll
    for (int v = 0; v < 8; ++v) {
      int i = t0 + hs * 8 + v;
      int j = s0 + n;
      c[ct][v] = is_masked(i, j) ? -1e30f : f.f[v] * ATTN_SCALE;
    }
  }

  // ---- row max (half-wave shuffle reduce, then cross-wave via LDS) ----
  #pragma unroll
  for (int v = 0; v < 8; ++v) {
    float m = c[0][v];
    #pragma unroll
    for (int ct = 1; ct < 12; ++ct) m = fmaxf(m, c[ct][v]);
    for (int off = 1; off < 16; off <<= 1) m = fmaxf(m, __shfl_xor(m, off, 32));
    if (n == 0) redmax[wave][hs * 8 + v] = m;
  }
  __syncthreads();
  float gmax[8];
  #pragma unroll
  for (int v = 0; v < 8; ++v) {
    float m = redmax[0][hs * 8 + v];
    #pragma unroll
    for (int w = 1; w < 8; ++w) m = fmaxf(m, redmax[w][hs * 8 + v]);
    gmax[v] = m;
  }

  // ---- exp and row sum ----
  #pragma unroll
  for (int v = 0; v < 8; ++v) {
    float s = 0.0f;
    #pragma unroll
    for (int ct = 0; ct < 12; ++ct) {
      float e = __expf(c[ct][v] - gmax[v]);
      c[ct][v] = e;
      s += e;
    }
    for (int off = 1; off < 16; off <<= 1) s += __shfl_xor(s, off, 32);
    if (n == 0) redsum[wave][hs * 8 + v] = s;
  }
  __syncthreads();
  float ginv[8];
  #pragma unroll
  for (int v = 0; v < 8; ++v) {
    float s = 0.0f;
    #pragma unroll
    for (int w = 0; w < 8; ++w) s += redsum[w][hs * 8 + v];
    ginv[v] = 1.0f / s;
  }

  // ---- normalize, stage P in LDS (bf16), accumulate attn mean over heads ----
  #pragma unroll
  for (int ct = 0; ct < 12; ++ct) {
    int s0 = (wave * 12 + ct) * 16;
    #pragma unroll
    for (int v = 0; v < 8; ++v) {
      int il = hs * 8 + v;
      int j = s0 + n;
      float p = c[ct][v] * ginv[v];
      Psh[il * T_LEN + j] = f2bf(p);
      atomicAdd(&attn_avg[((size_t)bb * T_LEN + (t0 + il)) * T_LEN + j],
                p * 0.0625f);
    }
  }
  __syncthreads();

  // ---- O_tile = P @ V : 4 output col-tiles x 2 K-halves over 8 waves ----
  int ctile = wave & 3, kh = wave >> 2;
  int kbeg = kh * (T_LEN / 2);
  v8f acc = {};
  for (int k = 0; k < T_LEN / 2; k += 32) {
    v16bf pa = load_a(Psh + kbeg + k, T_LEN, lane);
    v16bf vb = load_b(Vh + (size_t)ctile * 16 * T_LEN + kbeg + k, T_LEN, lane);
    acc = wmma_bf16(pa, vb, acc);
  }
  FragF f; f.v = acc;
  if (kh == 1) {
    #pragma unroll
    for (int v = 0; v < 8; ++v) accred[ctile * 256 + v * 32 + lane] = f.f[v];
  }
  __syncthreads();
  if (kh == 0) {
    int hh = bh & 15;
    #pragma unroll
    for (int v = 0; v < 8; ++v) {
      float o = f.f[v] + accred[ctile * 256 + v * 32 + lane];
      int t = t0 + hs * 8 + v;
      int row = t * NB + bb;                 // O is [T*B, E]
      int col = hh * HD + ctile * 16 + n;
      O[(size_t)row * EMB + col] = f2bf(o);
    }
  }
}

// ---------------------------------------------------------------------------
// Kernel 3: out = O @ out_w^T + out_b. M=6144, N=1024, K=1024.
// 32x32 register block per wave. fp32 output.
// ---------------------------------------------------------------------------
__global__ __launch_bounds__(256) void k_proj(
    const bf16_t* __restrict__ O, const bf16_t* __restrict__ OW,
    const float* __restrict__ ob, float* __restrict__ out) {
  int lane = threadIdx.x & 31, wave = threadIdx.x >> 5;
  int tile = blockIdx.x * 8 + wave;           // 768 blocks * 8 = 6144 tiles
  int mt = tile / 32, nt = tile % 32;         // 192 x 32 tiles of 32x32
  const bf16_t* A0 = O + (size_t)mt * 32 * EMB;
  const bf16_t* A1 = A0 + (size_t)16 * EMB;
  const bf16_t* B0 = OW + (size_t)nt * 32 * EMB;
  const bf16_t* B1 = B0 + (size_t)16 * EMB;
  v8f acc00 = {}, acc01 = {}, acc10 = {}, acc11 = {};
  for (int k = 0; k < EMB; k += 32) {
    v16bf a0 = load_a(A0 + k, EMB, lane);
    v16bf a1 = load_a(A1 + k, EMB, lane);
    v16bf b0 = load_b(B0 + k, EMB, lane);
    v16bf b1 = load_b(B1 + k, EMB, lane);
    acc00 = wmma_bf16(a0, b0, acc00);
    acc01 = wmma_bf16(a0, b1, acc01);
    acc10 = wmma_bf16(a1, b0, acc10);
    acc11 = wmma_bf16(a1, b1, acc11);
  }
  int n = lane & 15, hs = lane >> 4;
  #pragma unroll
  for (int si = 0; si < 2; ++si) {
    #pragma unroll
    for (int sj = 0; sj < 2; ++sj) {
      FragF f;
      f.v = si == 0 ? (sj == 0 ? acc00 : acc01) : (sj == 0 ? acc10 : acc11);
      int ng = nt * 32 + sj * 16 + n;
      float bv = ob[ng];
      #pragma unroll
      for (int v = 0; v < 8; ++v) {
        int m = mt * 32 + si * 16 + hs * 8 + v;
        out[(size_t)m * EMB + ng] = f.f[v] + bv;
      }
    }
  }
}

// ---------------------------------------------------------------------------
// Host launch
// ---------------------------------------------------------------------------
extern "C" void kernel_launch(void* const* d_in, const int* in_sizes, int n_in,
                              void* d_out, int out_size, void* d_ws, size_t ws_size,
                              hipStream_t stream) {
  const float* query = (const float*)d_in[0];   // [T, B, E] fp32
  const float* w_in  = (const float*)d_in[1];   // [3E, E]
  const float* b_in  = (const float*)d_in[2];   // [3E]
  const float* w_out = (const float*)d_in[3];   // [E, E]
  const float* b_out = (const float*)d_in[4];   // [E]
  float* out      = (float*)d_out;                          // [T, B, E]
  float* attn_avg = out + (size_t)MROWS * EMB;              // [B, T, T]

  // bf16 workspace layout (element offsets)
  bf16_t* ws = (bf16_t*)d_ws;
  bf16_t* X  = ws;                    // [6144, 1024]       6,291,456
  bf16_t* W  = X  + (size_t)6291456;  // [3072, 1024]       3,145,728
  bf16_t* OW = W  + (size_t)3145728;  // [1024, 1024]       1,048,576
  bf16_t* Q  = OW + (size_t)1048576;  // [B,H,T,64]         6,291,456
  bf16_t* K  = Q  + (size_t)6291456;  // [B,H,T,64]         6,291,456
  bf16_t* V  = K  + (size_t)6291456;  // [B,H,64,T] (V^T)   6,291,456
  bf16_t* O  = V  + (size_t)6291456;  // [6144, 1024]       6,291,456
  (void)in_sizes; (void)n_in; (void)out_size; (void)ws_size;

  k_convert<<<4096, 256, 0, stream>>>(query, w_in, w_out, X, W, OW);
  k_qkv<<<2304, 256, 0, stream>>>(X, W, b_in, Q, K, V);
  k_zero<<<4096, 256, 0, stream>>>(attn_avg, T_LEN * T_LEN * NB);
  k_attn<<<6144, 256, 0, stream>>>(Q, K, V, O, attn_avg);
  k_proj<<<768, 256, 0, stream>>>(O, OW, b_out, out);
}
